// MuleHunterGNN_49821620634268
// MI455X (gfx1250) — compile-verified, matrix operand order checked by reading, not run on hardware
//
#include <hip/hip_runtime.h>
#include <hip/hip_fp16.h>
#include <math.h>

#define N_NODES 50000
#define N_EDGES 1000000
#define IN_F    20
#define HID     64
#define HEADS   4
#define GDIM    (HEADS * HID)   // 256
#define OUT3    32
#define NCLS    2
#define EPS_BN  1e-5f

typedef __attribute__((ext_vector_type(16))) _Float16 v16h;
typedef __attribute__((ext_vector_type(8)))  float    v8f;

// ---------------- utility kernels ----------------

__global__ void fill_f32(float* __restrict__ p, float v, long n) {
    long i = (long)blockIdx.x * blockDim.x + threadIdx.x;
    if (i < n) p[i] = v;
}

__device__ __forceinline__ void atomicMaxF(float* addr, float v) {
    if (v >= 0.0f) atomicMax((int*)addr, __float_as_int(v));
    else           atomicMin((unsigned int*)addr, __float_as_uint(v));
}

__device__ __forceinline__ float bn_f(float x, float g, float b, float rm, float rv) {
    return (x - rm) * rsqrtf(rv + EPS_BN) * g + b;
}

// ---------------- skip branch: identity = x @ Wsk + bsk ----------------

__global__ void skip_kernel(const float* __restrict__ x, const float* __restrict__ Wsk,
                            const float* __restrict__ bsk, float* __restrict__ identity) {
    int tid = blockIdx.x * blockDim.x + threadIdx.x;
    if (tid >= N_NODES * OUT3) return;
    int n = tid >> 5, o = tid & 31;
    float acc = bsk[o];
    const float* xr = x + (size_t)n * IN_F;
#pragma unroll
    for (int j = 0; j < IN_F; ++j) acc += xr[j] * Wsk[j * OUT3 + o];
    identity[tid] = acc;
}

// ---------------- SAGE1: edge scatter (agg1 += x[src], cnt += 1) ----------------

__global__ void sage1_edge(const float* __restrict__ x, const long long* __restrict__ ei,
                           float* __restrict__ agg1, float* __restrict__ cnt) {
    int e = blockIdx.x * blockDim.x + threadIdx.x;
    if (e >= N_EDGES) return;
    long long src = ei[e], dst = ei[N_EDGES + e];
    atomicAdd(&cnt[dst], 1.0f);
    const float* xr = x + (size_t)src * IN_F;
    float* ar = agg1 + (size_t)dst * IN_F;
#pragma unroll
    for (int j = 0; j < IN_F; ++j) atomicAdd(&ar[j], xr[j]);
}

// ---------------- SAGE1 node: mean@Wl1 + x@Wr1 + b1, BN, ReLU -> f16 ----------------

__global__ void sage1_node(const float* __restrict__ x, const float* __restrict__ agg1,
                           const float* __restrict__ cnt,
                           const float* __restrict__ Wl1, const float* __restrict__ Wr1,
                           const float* __restrict__ b1,
                           const float* __restrict__ g1, const float* __restrict__ be1,
                           const float* __restrict__ rm1, const float* __restrict__ rv1,
                           _Float16* __restrict__ h1h) {
    int tid = blockIdx.x * blockDim.x + threadIdx.x;
    if (tid >= N_NODES * HID) return;
    int n = tid >> 6, o = tid & 63;
    float inv = 1.0f / fmaxf(cnt[n], 1.0f);
    const float* xr = x + (size_t)n * IN_F;
    const float* ar = agg1 + (size_t)n * IN_F;
    float acc = b1[o];
#pragma unroll
    for (int j = 0; j < IN_F; ++j)
        acc += (ar[j] * inv) * Wl1[j * HID + o] + xr[j] * Wr1[j * HID + o];
    float v = fmaxf(bn_f(acc, g1[o], be1[o], rm1[o], rv1[o]), 0.0f);
    h1h[tid] = (_Float16)v;
}

// ---------------- pack Wg (f32, 64x256 row-major) into WMMA B-fragment layout ------
// Fragment f = tN*2 + s (s = k0/32), 32 fragments total.
// Within fragment: lane l (m=l&15, hi=l>>4), element j in 0..15:
//   j<8 : k = k0 + 8*hi + j        col = tN*16 + m
//   j>=8: k = k0 + 16 + 8*hi + j-8
// Stored contiguously: Bp[((f*32 + l)*16 + j)] -> each lane reads 32B contiguous.

__global__ void pack_wg(const float* __restrict__ Wg, _Float16* __restrict__ Bp) {
    int tid = blockIdx.x * blockDim.x + threadIdx.x;
    if (tid >= 32 * 32 * 16) return;           // 16384 halves
    int f   = tid >> 9;                        // /512
    int rem = tid & 511;
    int l = rem >> 4, j = rem & 15;
    int tN = f >> 1, k0 = (f & 1) * 32;
    int m = l & 15, hi = l >> 4;
    int k = (j < 8) ? (k0 + 8 * hi + j) : (k0 + 16 + 8 * hi + (j - 8));
    Bp[tid] = (_Float16)Wg[k * GDIM + tN * 16 + m];
}

// ---------------- GAT GEMM: hg = h1 @ Wg  [50000x64]x[64x256] via WMMA -------------
// One wave per 16-row M-tile; A fragments loaded once and reused across all 16
// N-tiles (32 WMMAs per wave). B fragments are pre-packed -> contiguous b128 loads.

__global__ void gat_gemm_wmma(const _Float16* __restrict__ A, const _Float16* __restrict__ Bp,
                              float* __restrict__ C) {
    int wave = (int)((blockIdx.x * blockDim.x + threadIdx.x) >> 5);
    int lane = threadIdx.x & 31;
    if (wave >= N_NODES / 16) return;          // wave-uniform: EXEC all-1 for WMMA

    int m  = lane & 15;
    int hi = lane >> 4;
    const _Float16* Arow = A + (size_t)(wave * 16 + m) * HID;

    v16h a0, a1;
#pragma unroll
    for (int j = 0; j < 8; ++j) {              // contiguous 8-half runs -> b128 loads
        a0[j]     = Arow[8 * hi + j];
        a0[8 + j] = Arow[16 + 8 * hi + j];
        a1[j]     = Arow[32 + 8 * hi + j];
        a1[8 + j] = Arow[48 + 8 * hi + j];
    }

#pragma unroll
    for (int tN = 0; tN < 16; ++tN) {
        const v16h b0 = *(const v16h*)(Bp + ((size_t)(tN * 2 + 0) * 32 + lane) * 16);
        const v16h b1 = *(const v16h*)(Bp + ((size_t)(tN * 2 + 1) * 32 + lane) * 16);
        v8f acc = {};
        acc = __builtin_amdgcn_wmma_f32_16x16x32_f16(false, a0, false, b0, (short)0, acc,
                                                     false, false);
        acc = __builtin_amdgcn_wmma_f32_16x16x32_f16(false, a1, false, b1, (short)0, acc,
                                                     false, false);
        // C/D layout: VGPR r, lane l -> M = r + 8*hi, N = lane&15
#pragma unroll
        for (int r = 0; r < 8; ++r)
            C[(size_t)(wave * 16 + r + 8 * hi) * GDIM + tN * 16 + m] = acc[r];
    }
}

// ---------------- GAT attention coefficients per node/head ----------------

__global__ void attn_node(const float* __restrict__ hg,
                          const float* __restrict__ a_s, const float* __restrict__ a_d,
                          float* __restrict__ esrc, float* __restrict__ edstv) {
    int tid = blockIdx.x * blockDim.x + threadIdx.x;
    if (tid >= N_NODES * HEADS) return;
    int n = tid >> 2, h = tid & 3;
    const float* hr = hg + (size_t)n * GDIM + h * HID;
    const float* as = a_s + h * HID;
    const float* ad = a_d + h * HID;
    float s0 = 0.0f, s1 = 0.0f;
#pragma unroll
    for (int d = 0; d < HID; ++d) { s0 += hr[d] * as[d]; s1 += hr[d] * ad[d]; }
    esrc[tid] = s0;
    edstv[tid] = s1;
}

__device__ __forceinline__ float gat_logit(const float* esrc, const float* edstv,
                                           long long src, long long dst, int h) {
    float l = esrc[src * HEADS + h] + edstv[dst * HEADS + h];
    return l > 0.0f ? l : 0.2f * l;   // leaky_relu(0.2)
}

// ---------------- GAT edge pass 1: segment max of logits ----------------

__global__ void gat_edge_max(const long long* __restrict__ ei,
                             const float* __restrict__ esrc, const float* __restrict__ edstv,
                             float* __restrict__ mseg) {
    int e = blockIdx.x * blockDim.x + threadIdx.x;
    if (e >= N_EDGES) return;
    long long src = ei[e], dst = ei[N_EDGES + e];
#pragma unroll
    for (int h = 0; h < HEADS; ++h)
        atomicMaxF(&mseg[dst * HEADS + h], gat_logit(esrc, edstv, src, dst, h));
}

// ---------------- GAT edge pass 2: segment sum of exp(logit - m) ----------------

__global__ void gat_edge_sum(const long long* __restrict__ ei,
                             const float* __restrict__ esrc, const float* __restrict__ edstv,
                             const float* __restrict__ mseg, float* __restrict__ denom) {
    int e = blockIdx.x * blockDim.x + threadIdx.x;
    if (e >= N_EDGES) return;
    long long src = ei[e], dst = ei[N_EDGES + e];
#pragma unroll
    for (int h = 0; h < HEADS; ++h) {
        float ex = __expf(gat_logit(esrc, edstv, src, dst, h) - mseg[dst * HEADS + h]);
        atomicAdd(&denom[dst * HEADS + h], ex);
    }
}

// ---------------- GAT edge pass 3: weighted scatter of h[src] ----------------

__global__ void gat_edge_scatter(const long long* __restrict__ ei,
                                 const float* __restrict__ esrc, const float* __restrict__ edstv,
                                 const float* __restrict__ mseg, const float* __restrict__ denom,
                                 const float* __restrict__ hg, float* __restrict__ gacc) {
    int tid = blockIdx.x * blockDim.x + threadIdx.x;
    if (tid >= N_EDGES * HEADS) return;
    int e = tid >> 2, h = tid & 3;
    long long src = ei[e], dst = ei[N_EDGES + e];
    float ex = __expf(gat_logit(esrc, edstv, src, dst, h) - mseg[dst * HEADS + h]);
    float alpha = ex / fmaxf(denom[dst * HEADS + h], 1e-16f);
    const float4* hsrc = reinterpret_cast<const float4*>(hg + (size_t)src * GDIM + h * HID);
    float* acc = gacc + (size_t)dst * GDIM + h * HID;
#pragma unroll
    for (int q = 0; q < HID / 4; ++q) {
        float4 v = hsrc[q];
        atomicAdd(&acc[q * 4 + 0], alpha * v.x);
        atomicAdd(&acc[q * 4 + 1], alpha * v.y);
        atomicAdd(&acc[q * 4 + 2], alpha * v.z);
        atomicAdd(&acc[q * 4 + 3], alpha * v.w);
    }
}

// ---------------- GAT node: head-average + bias, BN, ReLU -> h2 ----------------

__global__ void gat_node(const float* __restrict__ gacc, const float* __restrict__ bg,
                         const float* __restrict__ g2, const float* __restrict__ be2,
                         const float* __restrict__ rm2, const float* __restrict__ rv2,
                         float* __restrict__ h2) {
    int tid = blockIdx.x * blockDim.x + threadIdx.x;
    if (tid >= N_NODES * HID) return;
    int n = tid >> 6, o = tid & 63;
    const float* gr = gacc + (size_t)n * GDIM;
    float v = 0.25f * (gr[o] + gr[HID + o] + gr[2 * HID + o] + gr[3 * HID + o]) + bg[o];
    h2[tid] = fmaxf(bn_f(v, g2[o], be2[o], rm2[o], rv2[o]), 0.0f);
}

// ---------------- SAGE3 edge scatter ----------------

__global__ void sage3_edge(const float* __restrict__ h2, const long long* __restrict__ ei,
                           float* __restrict__ agg2) {
    int e = blockIdx.x * blockDim.x + threadIdx.x;
    if (e >= N_EDGES) return;
    long long src = ei[e], dst = ei[N_EDGES + e];
    const float4* hr = reinterpret_cast<const float4*>(h2 + (size_t)src * HID);
    float* ar = agg2 + (size_t)dst * HID;
#pragma unroll
    for (int q = 0; q < HID / 4; ++q) {
        float4 v = hr[q];
        atomicAdd(&ar[q * 4 + 0], v.x);
        atomicAdd(&ar[q * 4 + 1], v.y);
        atomicAdd(&ar[q * 4 + 2], v.z);
        atomicAdd(&ar[q * 4 + 3], v.w);
    }
}

// ---------------- final fused node: SAGE3 + BN + ReLU + skip + MLP + log_softmax ----

__global__ void final_node(const float* __restrict__ h2, const float* __restrict__ agg2,
                           const float* __restrict__ cnt,
                           const float* __restrict__ Wl3, const float* __restrict__ Wr3,
                           const float* __restrict__ b3,
                           const float* __restrict__ g3, const float* __restrict__ be3,
                           const float* __restrict__ rm3, const float* __restrict__ rv3,
                           const float* __restrict__ identity,
                           const float* __restrict__ Wc1, const float* __restrict__ bc1,
                           const float* __restrict__ Wc2, const float* __restrict__ bc2,
                           float* __restrict__ out) {
    int n = blockIdx.x * blockDim.x + threadIdx.x;
    if (n >= N_NODES) return;
    float inv = 1.0f / fmaxf(cnt[n], 1.0f);
    const float* hr = h2 + (size_t)n * HID;
    const float* ar = agg2 + (size_t)n * HID;

    float emb[OUT3];
#pragma unroll
    for (int o = 0; o < OUT3; ++o) emb[o] = b3[o];
    for (int j = 0; j < HID; ++j) {
        float mj = ar[j] * inv, hj = hr[j];
#pragma unroll
        for (int o = 0; o < OUT3; ++o) emb[o] += mj * Wl3[j * OUT3 + o] + hj * Wr3[j * OUT3 + o];
    }
#pragma unroll
    for (int o = 0; o < OUT3; ++o) {
        float v = fmaxf(bn_f(emb[o], g3[o], be3[o], rm3[o], rv3[o]), 0.0f);
        emb[o] = v + identity[(size_t)n * OUT3 + o];
    }
    float c1[32];
#pragma unroll
    for (int o = 0; o < 32; ++o) {
        float a = bc1[o];
#pragma unroll
        for (int j = 0; j < OUT3; ++j) a += emb[j] * Wc1[j * 32 + o];
        c1[o] = fmaxf(a, 0.0f);
    }
    float l0 = bc2[0], l1 = bc2[1];
#pragma unroll
    for (int j = 0; j < 32; ++j) { l0 += c1[j] * Wc2[j * 2 + 0]; l1 += c1[j] * Wc2[j * 2 + 1]; }
    float mx = fmaxf(l0, l1);
    float lse = mx + __logf(__expf(l0 - mx) + __expf(l1 - mx));
    out[(size_t)n * NCLS + 0] = l0 - lse;
    out[(size_t)n * NCLS + 1] = l1 - lse;
}

// ---------------- launch ----------------

extern "C" void kernel_launch(void* const* d_in, const int* in_sizes, int n_in,
                              void* d_out, int out_size, void* d_ws, size_t ws_size,
                              hipStream_t stream) {
    const float*     x  = (const float*)d_in[0];
    const long long* ei = (const long long*)d_in[1];
    const float* Wl1 = (const float*)d_in[2];
    const float* Wr1 = (const float*)d_in[3];
    const float* b1  = (const float*)d_in[4];
    const float* g1  = (const float*)d_in[5];
    const float* be1 = (const float*)d_in[6];
    const float* rm1 = (const float*)d_in[7];
    const float* rv1 = (const float*)d_in[8];
    const float* Wg  = (const float*)d_in[9];
    const float* ags = (const float*)d_in[10];
    const float* agd = (const float*)d_in[11];
    const float* bg  = (const float*)d_in[12];
    const float* g2  = (const float*)d_in[13];
    const float* be2 = (const float*)d_in[14];
    const float* rm2 = (const float*)d_in[15];
    const float* rv2 = (const float*)d_in[16];
    const float* Wl3 = (const float*)d_in[17];
    const float* Wr3 = (const float*)d_in[18];
    const float* b3  = (const float*)d_in[19];
    const float* g3  = (const float*)d_in[20];
    const float* be3 = (const float*)d_in[21];
    const float* rm3 = (const float*)d_in[22];
    const float* rv3 = (const float*)d_in[23];
    const float* Wsk = (const float*)d_in[24];
    const float* bsk = (const float*)d_in[25];
    const float* Wc1 = (const float*)d_in[26];
    const float* bc1 = (const float*)d_in[27];
    const float* Wc2 = (const float*)d_in[28];
    const float* bc2 = (const float*)d_in[29];
    float* out = (float*)d_out;

    // workspace layout (256B aligned slices)
    char* w = (char*)d_ws;
    auto alloc = [&](size_t bytes) -> void* {
        void* p = (void*)w;
        w += (bytes + 255) & ~((size_t)255);
        return p;
    };
    float*     identity = (float*)alloc((size_t)N_NODES * OUT3 * 4);
    float*     agg1     = (float*)alloc((size_t)N_NODES * IN_F * 4);
    float*     cnt      = (float*)alloc((size_t)N_NODES * 4);
    _Float16*  h1h      = (_Float16*)alloc((size_t)N_NODES * HID * 2);
    _Float16*  Bpack    = (_Float16*)alloc((size_t)32 * 32 * 16 * 2);   // 32 fragments
    float*     hg       = (float*)alloc((size_t)N_NODES * GDIM * 4);
    float*     esrc     = (float*)alloc((size_t)N_NODES * HEADS * 4);
    float*     edstv    = (float*)alloc((size_t)N_NODES * HEADS * 4);
    float*     mseg     = (float*)alloc((size_t)N_NODES * HEADS * 4);
    float*     denom    = (float*)alloc((size_t)N_NODES * HEADS * 4);
    float*     gacc     = (float*)alloc((size_t)N_NODES * GDIM * 4);
    float*     h2       = (float*)alloc((size_t)N_NODES * HID * 4);
    float*     agg2     = hg;   // hg is dead after gat_edge_scatter; reuse for SAGE3 agg

    const int T = 256;
    auto blocks = [&](long n) { return (unsigned)((n + T - 1) / T); };

    // init accumulators (graph-replayed every call)
    fill_f32<<<blocks((long)N_NODES * IN_F), T, 0, stream>>>(agg1, 0.0f, (long)N_NODES * IN_F);
    fill_f32<<<blocks(N_NODES), T, 0, stream>>>(cnt, 0.0f, N_NODES);
    fill_f32<<<blocks((long)N_NODES * HEADS), T, 0, stream>>>(mseg, -INFINITY, (long)N_NODES * HEADS);
    fill_f32<<<blocks((long)N_NODES * HEADS), T, 0, stream>>>(denom, 0.0f, (long)N_NODES * HEADS);
    fill_f32<<<blocks((long)N_NODES * GDIM), T, 0, stream>>>(gacc, 0.0f, (long)N_NODES * GDIM);

    // skip branch + weight pack (independent)
    skip_kernel<<<blocks((long)N_NODES * OUT3), T, 0, stream>>>(x, Wsk, bsk, identity);
    pack_wg<<<blocks(32 * 32 * 16), T, 0, stream>>>(Wg, Bpack);

    // SAGE1
    sage1_edge<<<blocks(N_EDGES), T, 0, stream>>>(x, ei, agg1, cnt);
    sage1_node<<<blocks((long)N_NODES * HID), T, 0, stream>>>(x, agg1, cnt, Wl1, Wr1, b1,
                                                             g1, be1, rm1, rv1, h1h);

    // GAT: WMMA GEMM (one wave per M-tile, 32 WMMAs/wave) -> attention -> 3 edge passes
    {
        long waves = N_NODES / 16;              // 3125
        gat_gemm_wmma<<<blocks(waves * 32), T, 0, stream>>>(h1h, Bpack, hg);
    }
    attn_node<<<blocks((long)N_NODES * HEADS), T, 0, stream>>>(hg, ags, agd, esrc, edstv);
    gat_edge_max<<<blocks(N_EDGES), T, 0, stream>>>(ei, esrc, edstv, mseg);
    gat_edge_sum<<<blocks(N_EDGES), T, 0, stream>>>(ei, esrc, edstv, mseg, denom);
    gat_edge_scatter<<<blocks((long)N_EDGES * HEADS), T, 0, stream>>>(ei, esrc, edstv, mseg,
                                                                     denom, hg, gacc);
    gat_node<<<blocks((long)N_NODES * HID), T, 0, stream>>>(gacc, bg, g2, be2, rm2, rv2, h2);

    // SAGE3 (agg2 aliases hg, zero it now that hg is dead)
    fill_f32<<<blocks((long)N_NODES * HID), T, 0, stream>>>(agg2, 0.0f, (long)N_NODES * HID);
    sage3_edge<<<blocks(N_EDGES), T, 0, stream>>>(h2, ei, agg2);

    // fused SAGE3 + BN + ReLU + skip + classifier + log_softmax
    final_node<<<blocks(N_NODES), T, 0, stream>>>(h2, agg2, cnt, Wl3, Wr3, b3,
                                                  g3, be3, rm3, rv3, identity,
                                                  Wc1, bc1, Wc2, bc2, out);
}
